// RJCausalMultiHeadSelfAttention_3521873183686
// MI455X (gfx1250) — matrix operand order repaired and checked by reading, not compiled, not run on hardware
//
#include <hip/hip_runtime.h>
#include <hip/hip_bf16.h>
#include <math.h>

typedef __attribute__((ext_vector_type(16))) _Float16 v16h;
typedef __attribute__((ext_vector_type(8)))  float    v8f;
typedef __attribute__((ext_vector_type(4)))  float    v4f;
typedef __attribute__((ext_vector_type(4)))  unsigned int v4u;
typedef __attribute__((ext_vector_type(8)))  int      v8i;
typedef __attribute__((ext_vector_type(4)))  int      v4i;

#define DMODEL 1024
#define NHEADS 16
#define DK     64
#define SEQ    2048
#define BATCH  4
#define MS     (BATCH*SEQ)   /* 8192 rows */

union FragU { v16h h; v4f f[2]; };
union H4   { uint2 u; _Float16 h[4]; };

#if __has_builtin(__builtin_amdgcn_tensor_load_to_lds)
#define HAVE_TDM 1
#else
#define HAVE_TDM 0
#endif

__device__ __forceinline__ v8f wmma16(v16h a, v16h b, v8f c) {
  // D = A(16x32 f16) x B(32x16 f16) + C(16x16 f32)
  return __builtin_amdgcn_wmma_f32_16x16x32_f16(
      /*neg_a=*/false, a, /*neg_b=*/false, b,
      /*c_mod=*/(short)0, c, /*reuse_a=*/false, /*reuse_b=*/false);
}

// A-matrix fragment: 16-bit A 16x32 layout (ISA 7.12.2).
// lanes 0-15: V0..3 = K0..7, V4..7 = K16..23 ; lanes 16-31: K8..15, K24..31
__device__ __forceinline__ v16h load_fragA(const _Float16* base, int rowByte, int hlf) {
  const char* p = (const char*)base + rowByte;
  FragU u;
  u.f[0] = *(const v4f*)(p + 16*hlf);
  u.f[1] = *(const v4f*)(p + 32 + 16*hlf);
  return u.h;
}

// B-matrix fragment: same K order as A fragment:
// lanes 0-15: K0..7, K16..23 ; lanes 16-31: K8..15, K24..31.
__device__ __forceinline__ v16h load_fragB(const _Float16* base, int rowByte, int hlf) {
  const char* p = (const char*)base + rowByte;
  FragU u;
  u.f[0] = *(const v4f*)(p + 16*hlf);
  u.f[1] = *(const v4f*)(p + 32 + 16*hlf);
  return u.h;
}

#if HAVE_TDM
// ---- Tensor Data Mover descriptor construction (ISA 08_async_tensor §8) ----
union G0u { unsigned long long q[2]; v4u v; };
union G1u { unsigned long long q[4]; v8i v; };

// Group0: count=1, lds_addr[63:32], global_addr[120:64], type=2 at [127:126]
__device__ __forceinline__ v4u tdm_g0(unsigned lds_addr, unsigned long long gaddr) {
  G0u g;
  g.q[0] = 1ull | ((unsigned long long)lds_addr << 32);
  g.q[1] = (gaddr & 0x01FFFFFFFFFFFFFFull) | (2ull << 62);
  return g.v;
}
// Group1: data_size=1 (2B) [17:16], tensor_dim0 [79:48], tensor_dim1 [111:80],
// tile_dim0 [127:112], tile_dim1 [143:128], tensor_dim0_stride [207:160]
__device__ __forceinline__ v8i tdm_g1(unsigned tdim0, unsigned tdim1,
                                      unsigned tile0, unsigned tile1,
                                      unsigned long long stride0) {
  G1u g;
  g.q[0] = (1ull << 16) | ((unsigned long long)(tdim0 & 0xFFFFu) << 48);
  g.q[1] = (unsigned long long)(tdim0 >> 16)
         | ((unsigned long long)tdim1 << 16)
         | ((unsigned long long)(tile0 & 0xFFFFu) << 48);
  g.q[2] = (unsigned long long)(tile1 & 0xFFFFu)
         | ((stride0 & 0xFFFFFFFFull) << 32);
  g.q[3] = (stride0 >> 32) & 0xFFFFull;   // tensor_dim1_stride = 0 (2D)
  return g.v;
}

__device__ __forceinline__ void tdm_load(v4u g0, v8i g1) {
  v4i z4 = {};
#if defined(__clang_major__) && __clang_major__ >= 23
  v8i z8 = {};
  __builtin_amdgcn_tensor_load_to_lds(g0, g1, z4, z4, z8, 0);
#else
  __builtin_amdgcn_tensor_load_to_lds(g0, g1, z4, z4, 0);
#endif
}

__device__ __forceinline__ unsigned lds_byte_off(const void* p) {
  // generic LDS address carries the LDS byte offset in its low 32 bits
  return (unsigned)(unsigned long long)(uintptr_t)p;
}
#endif  // HAVE_TDM

__global__ __launch_bounds__(256)
void cvt_f32_f16(const float* __restrict__ src, _Float16* __restrict__ dst, int n) {
  int i = (blockIdx.x * 256 + threadIdx.x) * 4;
  if (i >= n) return;
  v4f x = *(const v4f*)(src + i);
  H4 o;
  o.h[0] = (_Float16)x.x; o.h[1] = (_Float16)x.y;
  o.h[2] = (_Float16)x.z; o.h[3] = (_Float16)x.w;
  *(uint2*)(dst + i) = o.u;
}

// C[M,N] = A[M,K] @ B[N,K]^T   (A,B f16 row-major, C = f16 or f32)
template <typename OutT>
__global__ __launch_bounds__(256)
void gemm_xt_wmma(const _Float16* __restrict__ A, const _Float16* __restrict__ B,
                  OutT* __restrict__ C, int M, int N, int K) {
  __shared__ _Float16 Ash[128 * 32];
  __shared__ _Float16 Bsh[64 * 32];
  const int tid  = threadIdx.x;
  const int lane = tid & 31;
  const int w    = tid >> 5;
  const int hlf  = lane >> 4;
  const int ln   = lane & 15;
  const int wm   = w & 3;      // 4 M-subtiles of 32
  const int wn   = w >> 2;     // 2 N-subtiles of 32
  const int m0   = blockIdx.y * 128;
  const int n0   = blockIdx.x * 64;

  v8f acc[2][2] = {};

#if HAVE_TDM
  // Invariant descriptor parts: 128x32 / 64x32 f16 tiles, row stride = K elems
  const v8i g1A = tdm_g1((unsigned)K, (unsigned)M, 32u, 128u, (unsigned long long)K);
  const v8i g1B = tdm_g1((unsigned)K, (unsigned)N, 32u, 64u,  (unsigned long long)K);
  const unsigned ldsA = lds_byte_off(&Ash[0]);
  const unsigned ldsB = lds_byte_off(&Bsh[0]);
  const unsigned long long gA = (unsigned long long)(uintptr_t)A;
  const unsigned long long gB = (unsigned long long)(uintptr_t)B;
#endif

  for (int k0 = 0; k0 < K; k0 += 32) {
    __syncthreads();
#if HAVE_TDM
    if (w == 0) {
      // TDM DMA: global tile -> LDS, tracked by TENSORcnt
      tdm_load(tdm_g0(ldsA, gA + ((size_t)m0 * K + k0) * 2ull), g1A);
      tdm_load(tdm_g0(ldsB, gB + ((size_t)n0 * K + k0) * 2ull), g1B);
      __builtin_amdgcn_s_wait_tensorcnt(0);
    }
#else
    for (int i = tid; i < 128 * 8; i += 256) {   // 128x32 f16 tile, 8B chunks
      int r = i >> 3, cc = i & 7;
      *(uint2*)&Ash[r * 32 + cc * 4] =
          *(const uint2*)&A[(size_t)(m0 + r) * K + k0 + cc * 4];
    }
    for (int i = tid; i < 64 * 8; i += 256) {    // 64x32 f16 tile
      int r = i >> 3, cc = i & 7;
      *(uint2*)&Bsh[r * 32 + cc * 4] =
          *(const uint2*)&B[(size_t)(n0 + r) * K + k0 + cc * 4];
    }
#endif
    if (k0 + 32 < K)  // warm next K tile into cache (global_prefetch_b8)
      __builtin_prefetch(&A[(size_t)(m0 + (tid >> 1)) * K + k0 + 32], 0, 0);
    __syncthreads();

    v16h a0 = load_fragA(Ash, (wm * 32 + ln) * 64, hlf);
    v16h a1 = load_fragA(Ash, (wm * 32 + 16 + ln) * 64, hlf);
    v16h b0 = load_fragB(Bsh, (wn * 32 + ln) * 64, hlf);
    v16h b1 = load_fragB(Bsh, (wn * 32 + 16 + ln) * 64, hlf);
    acc[0][0] = wmma16(a0, b0, acc[0][0]);
    acc[0][1] = wmma16(a0, b1, acc[0][1]);
    acc[1][0] = wmma16(a1, b0, acc[1][0]);
    acc[1][1] = wmma16(a1, b1, acc[1][1]);
  }

  for (int mi = 0; mi < 2; ++mi)
    for (int ni = 0; ni < 2; ++ni)
      for (int v = 0; v < 8; ++v) {
        int row = m0 + wm * 32 + mi * 16 + v + 8 * hlf;
        int col = n0 + wn * 32 + ni * 16 + ln;
        C[(size_t)row * N + col] = (OutT)acc[mi][ni][v];
      }
}

__global__ __launch_bounds__(256)
void rope_kernel(_Float16* __restrict__ Q, _Float16* __restrict__ Kc,
                 const int* __restrict__ pos, int npairs) {
  int idx = blockIdx.x * 256 + threadIdx.x;
  if (idx >= npairs) return;
  int m    = idx >> 9;          // row in [B*S, D] (512 pairs per row)
  int pr   = idx & 511;
  int col  = pr * 2;
  int kidx = pr & 31;           // (col % 64) / 2
  float p  = (float)pos[m];
  const float lnTheta = 9.210340371976184f;  // ln(10000)
  float invf = __expf(-(2.0f * (float)kidx / (float)DK) * lnTheta);
  float ang  = p * invf;
  float sn, cs;
  sincosf(ang, &sn, &cs);
  size_t off = (size_t)m * DMODEL + col;
  float qe = (float)Q[off], qo = (float)Q[off + 1];
  Q[off]     = (_Float16)(qe * cs - qo * sn);
  Q[off + 1] = (_Float16)(qe * sn + qo * cs);
  float ke = (float)Kc[off], ko = (float)Kc[off + 1];
  Kc[off]     = (_Float16)(ke * cs - ko * sn);
  Kc[off + 1] = (_Float16)(ke * sn + ko * cs);
}

// Flash attention: grid (S/128, B*H). Block = 8 waves, wave owns 16 q rows.
__global__ __launch_bounds__(256)
void attn_kernel(const _Float16* __restrict__ Qg, const _Float16* __restrict__ Kg,
                 const _Float16* __restrict__ Vg, _Float16* __restrict__ Og) {
  __shared__ _Float16 Qs[128 * 64];
  __shared__ _Float16 Ks[64 * 64];
  __shared__ _Float16 Vt[64 * 64];        // transposed: Vt[d][k]
  __shared__ _Float16 Ps[8][16 * 64];     // per-wave P scratch

  const int tid  = threadIdx.x;
  const int lane = tid & 31;
  const int w    = tid >> 5;
  const int hlf  = lane >> 4;
  const int ln   = lane & 15;
  const int qb   = blockIdx.x;
  const int b    = blockIdx.y >> 4;
  const int h    = blockIdx.y & 15;
  const size_t rowQ0 = (size_t)(b * SEQ + qb * 128);

  for (int i = tid; i < 128 * 16; i += 256) {
    int r = i >> 4, cc = i & 15;
    *(uint2*)&Qs[r * 64 + cc * 4] =
        *(const uint2*)&Qg[(rowQ0 + r) * DMODEL + h * DK + cc * 4];
  }
  __syncthreads();

  v16h qf0 = load_fragA(Qs, (w * 16 + ln) * 128, hlf);        // d 0..31
  v16h qf1 = load_fragA(Qs, (w * 16 + ln) * 128 + 64, hlf);   // d 32..63

  float mrun[8], lrun[8];
  v8f o[4] = {};
  for (int v = 0; v < 8; ++v) { mrun[v] = -INFINITY; lrun[v] = 0.f; }

  const int kblocks = 2 * qb + 2;                  // causal: k <= qb*128+127
  const int qrow0   = qb * 128 + w * 16 + 8 * hlf; // rows qrow0+v

  for (int kb0 = 0; kb0 < kblocks; ++kb0) {
    const int kb = kb0 * 64;
    __syncthreads();
    for (int i = tid; i < 64 * 16; i += 256) {
      int r = i >> 4, cc = i & 15;
      size_t gro = (size_t)(b * SEQ + kb + r) * DMODEL + h * DK + cc * 4;
      *(uint2*)&Ks[r * 64 + cc * 4] = *(const uint2*)&Kg[gro];
      H4 t; t.u = *(const uint2*)&Vg[gro];
      Vt[(cc * 4 + 0) * 64 + r] = t.h[0];
      Vt[(cc * 4 + 1) * 64 + r] = t.h[1];
      Vt[(cc * 4 + 2) * 64 + r] = t.h[2];
      Vt[(cc * 4 + 3) * 64 + r] = t.h[3];
    }
    __syncthreads();

    // S = Q @ K^T  (16 x 64 per wave)
    v8f sc[4] = {};
    for (int ct = 0; ct < 4; ++ct) {
      v16h bf0 = load_fragB(Ks, (ct * 16 + ln) * 128, hlf);
      v16h bf1 = load_fragB(Ks, (ct * 16 + ln) * 128 + 64, hlf);
      sc[ct] = wmma16(qf0, bf0, sc[ct]);
      sc[ct] = wmma16(qf1, bf1, sc[ct]);
    }

    // causal mask + scale + tile row-max
    float tmax[8];
    for (int v = 0; v < 8; ++v) tmax[v] = -INFINITY;
    for (int ct = 0; ct < 4; ++ct) {
      int kc = kb + ct * 16 + ln;
      for (int v = 0; v < 8; ++v) {
        float s = sc[ct][v] * 0.125f;          // 1/sqrt(64)
        s = (kc <= qrow0 + v) ? s : -INFINITY;
        sc[ct][v] = s;
        tmax[v] = fmaxf(tmax[v], s);
      }
    }
    for (int off = 1; off < 16; off <<= 1)
      for (int v = 0; v < 8; ++v)
        tmax[v] = fmaxf(tmax[v], __shfl_xor(tmax[v], off, 32));

    float cf[8], tsum[8];
    for (int v = 0; v < 8; ++v) {
      float mnew = fmaxf(mrun[v], tmax[v]);
      cf[v]   = __expf(mrun[v] - mnew);        // 0 on first block
      mrun[v] = mnew;
      tsum[v] = 0.f;
    }
    for (int ct = 0; ct < 4; ++ct)
      for (int v = 0; v < 8; ++v) {
        float pv = __expf(sc[ct][v] - mrun[v]);
        tsum[v] += pv;
        Ps[w][(v + 8 * hlf) * 64 + ct * 16 + ln] = (_Float16)pv;
      }
    for (int off = 1; off < 16; off <<= 1)
      for (int v = 0; v < 8; ++v)
        tsum[v] += __shfl_xor(tsum[v], off, 32);
    for (int v = 0; v < 8; ++v) lrun[v] = lrun[v] * cf[v] + tsum[v];
    for (int dt = 0; dt < 4; ++dt)
      for (int v = 0; v < 8; ++v) o[dt][v] *= cf[v];

    // O += P @ V   (LDS ops are in-order per wave: Ps write->read is safe)
    for (int c = 0; c < 2; ++c) {
      v16h pf = load_fragA(&Ps[w][0], ln * 128 + c * 64, hlf);
      for (int dt = 0; dt < 4; ++dt) {
        v16h vf = load_fragB(Vt, (dt * 16 + ln) * 128 + c * 64, hlf);
        o[dt] = wmma16(pf, vf, o[dt]);
      }
    }
  }

  for (int dt = 0; dt < 4; ++dt)
    for (int v = 0; v < 8; ++v) {
      size_t grow = rowQ0 + w * 16 + v + 8 * hlf;
      int    gcol = h * DK + dt * 16 + ln;
      Og[grow * DMODEL + gcol] = (_Float16)(o[dt][v] / lrun[v]);
    }
}

extern "C" void kernel_launch(void* const* d_in, const int* in_sizes, int n_in,
                              void* d_out, int out_size, void* d_ws, size_t ws_size,
                              hipStream_t stream) {
  const float* X   = (const float*)d_in[0];
  const int*   pos = (const int*)d_in[1];   // JAX x64-off: int64 -> int32
  const float* Wq  = (const float*)d_in[2];
  const float* Wk  = (const float*)d_in[3];
  const float* Wv  = (const float*)d_in[4];
  const float* Wo  = (const float*)d_in[5];
  float* out = (float*)d_out;

  _Float16* Xh  = (_Float16*)d_ws;
  _Float16* Wqh = Xh  + (size_t)MS * DMODEL;
  _Float16* Wkh = Wqh + (size_t)DMODEL * DMODEL;
  _Float16* Wvh = Wkh + (size_t)DMODEL * DMODEL;
  _Float16* Woh = Wvh + (size_t)DMODEL * DMODEL;
  _Float16* Qh  = Woh + (size_t)DMODEL * DMODEL;
  _Float16* Kh  = Qh  + (size_t)MS * DMODEL;
  _Float16* Vh  = Kh  + (size_t)MS * DMODEL;
  _Float16* Ah  = Vh  + (size_t)MS * DMODEL;

  cvt_f32_f16<<<MS * DMODEL / 1024, 256, 0, stream>>>(X, Xh, MS * DMODEL);
  cvt_f32_f16<<<DMODEL * DMODEL / 1024, 256, 0, stream>>>(Wq, Wqh, DMODEL * DMODEL);
  cvt_f32_f16<<<DMODEL * DMODEL / 1024, 256, 0, stream>>>(Wk, Wkh, DMODEL * DMODEL);
  cvt_f32_f16<<<DMODEL * DMODEL / 1024, 256, 0, stream>>>(Wv, Wvh, DMODEL * DMODEL);
  cvt_f32_f16<<<DMODEL * DMODEL / 1024, 256, 0, stream>>>(Wo, Woh, DMODEL * DMODEL);

  dim3 gg(DMODEL / 64, MS / 128);
  gemm_xt_wmma<_Float16><<<gg, 256, 0, stream>>>(Xh, Wqh, Qh, MS, DMODEL, DMODEL);
  gemm_xt_wmma<_Float16><<<gg, 256, 0, stream>>>(Xh, Wkh, Kh, MS, DMODEL, DMODEL);
  gemm_xt_wmma<_Float16><<<gg, 256, 0, stream>>>(Xh, Wvh, Vh, MS, DMODEL, DMODEL);

  rope_kernel<<<MS * DMODEL / 2 / 256, 256, 0, stream>>>(Qh, Kh, pos, MS * DMODEL / 2);

  dim3 ag(SEQ / 128, BATCH * NHEADS);
  attn_kernel<<<ag, 256, 0, stream>>>(Qh, Kh, Vh, Ah);

  gemm_xt_wmma<float><<<gg, 256, 0, stream>>>(Ah, Woh, out, MS, DMODEL, DMODEL);
}
